// ChannelAttentionBlock_3324304687557
// MI455X (gfx1250) — matrix operand order repaired and checked by reading, not compiled, not run on hardware
//
#include <hip/hip_runtime.h>
#include <hip/hip_bf16.h>
#include <math.h>

// ---------------------------------------------------------------------------
// ChannelAttentionBlock for MI455X (gfx1250, wave32, WMMA).
//   B=8, C=6, HW=4096, FF=9216
// Stage 1: h = relu(feature[48,9216] @ fc1_w^T + b1)      (bf16 WMMA, f32 acc)
// Stage 2: f = h[48,9216] @ fc2_w^T + b2 -> [48,4096]     (bf16 WMMA, f32 acc)
// Stage 3: per-(b,m) softmax stats over n of ls[m]*dot6(x[:,n], f[:,m])
// Stage 4: out[b,n,c] = sum_m softmaxed(temp)[n,m]*x[b,c,m]; LayerNorm(C=6);
//          write transposed [B,C,HW].  temp (537MB) is never materialized.
// ---------------------------------------------------------------------------

typedef __attribute__((ext_vector_type(16))) __bf16 v16bf;
typedef __attribute__((ext_vector_type(8)))  float  v8f;

#define NB 8
#define NC 6
#define NHW 4096
#define NFF 9216
#define NR (NB * NC)        // 48 rows = 3 WMMA M-tiles

// ---- tile loaders: fp32 global -> bf16 WMMA operand layout (wave32) --------
// A 16x32 bf16: lane holds row L%16; 16 contiguous K at (L/16)*16.
__device__ inline v16bf load_a_tile(const float* __restrict__ p) {
  const float4* q = reinterpret_cast<const float4*>(p);
  float4 a0 = q[0], a1 = q[1], a2 = q[2], a3 = q[3]; // +0,+4,+8,+12 floats
  v16bf r;
  r[0]=(__bf16)a0.x; r[1]=(__bf16)a0.y; r[2]=(__bf16)a0.z; r[3]=(__bf16)a0.w;
  r[4]=(__bf16)a1.x; r[5]=(__bf16)a1.y; r[6]=(__bf16)a1.z; r[7]=(__bf16)a1.w;
  r[8]=(__bf16)a2.x; r[9]=(__bf16)a2.y; r[10]=(__bf16)a2.z; r[11]=(__bf16)a2.w;
  r[12]=(__bf16)a3.x; r[13]=(__bf16)a3.y; r[14]=(__bf16)a3.z; r[15]=(__bf16)a3.w;
  return r;
}
// B 32x16 bf16: lane holds column L%16; 16 contiguous K at (L/16)*16.
__device__ inline v16bf load_b_tile(const float* __restrict__ p) {
  const float4* q = reinterpret_cast<const float4*>(p);
  float4 a0 = q[0], a1 = q[1], a2 = q[2], a3 = q[3];
  v16bf r;
  r[0]=(__bf16)a0.x; r[1]=(__bf16)a0.y; r[2]=(__bf16)a0.z; r[3]=(__bf16)a0.w;
  r[4]=(__bf16)a1.x; r[5]=(__bf16)a1.y; r[6]=(__bf16)a1.z; r[7]=(__bf16)a1.w;
  r[8]=(__bf16)a2.x; r[9]=(__bf16)a2.y; r[10]=(__bf16)a2.z; r[11]=(__bf16)a2.w;
  r[12]=(__bf16)a3.x; r[13]=(__bf16)a3.y; r[14]=(__bf16)a3.z; r[15]=(__bf16)a3.w;
  return r;
}

// ---- skinny GEMM: O[48,N] = act(A[48,K] @ W[N,K]^T + bias) -----------------
// Block = 128 threads (4 waves). Block owns one 16-wide N strip; each wave
// takes K/4 of the reduction; partial C tiles reduced through LDS.
template <bool RELU>
__global__ void __launch_bounds__(128)
gemm48_kernel(const float* __restrict__ A, const float* __restrict__ W,
              const float* __restrict__ bias, float* __restrict__ O,
              int K, int N) {
  const int wave  = threadIdx.x >> 5;
  const int lane  = threadIdx.x & 31;
  const int l16   = lane & 15;
  const int lhalf = lane >> 4;
  const int n0    = blockIdx.x << 4;

  const int Kw   = K >> 2;           // K per wave (9216/4 = 2304; 72 k-steps)
  const int kbeg = wave * Kw;
  const int kend = kbeg + Kw;

  v8f c0 = {}, c1 = {}, c2 = {};

  const float* pB0 = W + (size_t)(n0 + l16) * K + lhalf * 16;
  const float* pA0 = A + (size_t)l16 * K + lhalf * 16;

  for (int k0 = kbeg; k0 < kend; k0 += 32) {
    const float* pB = pB0 + k0;
    __builtin_prefetch(pB + 32, 0, 1);          // global_prefetch_b8 next tile
    v16bf bt = load_b_tile(pB);

    v16bf a0 = load_a_tile(pA0 + k0);
    v16bf a1 = load_a_tile(pA0 + (size_t)16 * K + k0);
    v16bf a2 = load_a_tile(pA0 + (size_t)32 * K + k0);

    c0 = __builtin_amdgcn_wmma_f32_16x16x32_bf16(false, a0, false, bt,
                                                 (short)0, c0, false, false);
    c1 = __builtin_amdgcn_wmma_f32_16x16x32_bf16(false, a1, false, bt,
                                                 (short)0, c1, false, false);
    c2 = __builtin_amdgcn_wmma_f32_16x16x32_bf16(false, a2, false, bt,
                                                 (short)0, c2, false, false);
  }

  // C layout: VGPR d -> row d + lhalf*8, col l16.
  __shared__ float red[4][3][256];
#pragma unroll
  for (int d = 0; d < 8; ++d) {
    const int row = d + lhalf * 8;
    red[wave][0][row * 16 + l16] = c0[d];
    red[wave][1][row * 16 + l16] = c1[d];
    red[wave][2][row * 16 + l16] = c2[d];
  }
  __syncthreads();

  for (int i = threadIdx.x; i < 3 * 256; i += 128) {
    const int mt = i >> 8, rc = i & 255;
    float s = red[0][mt][rc] + red[1][mt][rc] + red[2][mt][rc] + red[3][mt][rc];
    const int row = mt * 16 + (rc >> 4), col = rc & 15;
    s += bias[n0 + col];
    if (RELU) s = fmaxf(s, 0.0f);
    O[(size_t)row * N + n0 + col] = s;
  }
}

// ---- Stage 3: per-(b,m) online softmax stats over n ------------------------
// temp[b,n,m] = ls[m] * sum_c x[b,c,n] * f[b,c,m];  stats over n (axis=1).
__global__ void __launch_bounds__(256)
colstats_kernel(const float* __restrict__ x, const float* __restrict__ f,
                const float* __restrict__ logit_scale,
                float* __restrict__ colmax, float* __restrict__ colinv,
                float* __restrict__ lsv) {
  const int b = blockIdx.x >> 4;
  const int m = ((blockIdx.x & 15) << 8) + threadIdx.x;

  float fr[NC];
#pragma unroll
  for (int c = 0; c < NC; ++c) fr[c] = f[((size_t)(b * NC + c)) * NHW + m];

  const float ls = __expf(fminf(logit_scale[m], 4.6051702f)); // log(100)
  if (b == 0) lsv[m] = ls;

  float mx = -INFINITY, sm = 0.0f;
  __shared__ float xs[NC][512];
  for (int nb = 0; nb < NHW; nb += 512) {
    __syncthreads();
    for (int i = threadIdx.x; i < NC * 512; i += 256) {
      const int c = i >> 9, n = i & 511;
      xs[c][n] = x[((size_t)(b * NC + c)) * NHW + nb + n];
    }
    __syncthreads();
#pragma unroll 4
    for (int n = 0; n < 512; ++n) {
      float v = 0.0f;
#pragma unroll
      for (int c = 0; c < NC; ++c) v = fmaf(xs[c][n], fr[c], v);
      v *= ls;
      if (v <= mx) {
        sm += __expf(v - mx);
      } else {
        sm = sm * __expf(mx - v) + 1.0f;
        mx = v;
      }
    }
  }
  colmax[b * NHW + m] = mx;
  colinv[b * NHW + m] = 1.0f / sm;
}

// ---- Stage 4: out accumulate + LayerNorm(C=6) + transpose ------------------
__global__ void __launch_bounds__(256)
attnout_kernel(const float* __restrict__ x, const float* __restrict__ f,
               const float* __restrict__ colmax, const float* __restrict__ colinv,
               const float* __restrict__ lsv, const float* __restrict__ norm_w,
               const float* __restrict__ norm_b, float* __restrict__ out) {
  const int b = blockIdx.x >> 4;
  const int n = ((blockIdx.x & 15) << 8) + threadIdx.x;

  float xr[NC];
#pragma unroll
  for (int c = 0; c < NC; ++c) xr[c] = x[((size_t)(b * NC + c)) * NHW + n];

  float acc[NC] = {0.f, 0.f, 0.f, 0.f, 0.f, 0.f};

  __shared__ float fs[NC][256];
  __shared__ float xs[NC][256];
  __shared__ float cmx[256], cin[256], lss[256];

  for (int mb = 0; mb < NHW; mb += 256) {
    __syncthreads();
    {
      const int t = threadIdx.x;
      for (int i = t; i < NC * 256; i += 256) {
        const int c = i >> 8, mm = i & 255;
        fs[c][mm] = f[((size_t)(b * NC + c)) * NHW + mb + mm];
        xs[c][mm] = x[((size_t)(b * NC + c)) * NHW + mb + mm];
      }
      cmx[t] = colmax[b * NHW + mb + t];
      cin[t] = colinv[b * NHW + mb + t];
      lss[t] = lsv[mb + t];
    }
    __syncthreads();
#pragma unroll 4
    for (int m = 0; m < 256; ++m) {
      float v = 0.0f;
#pragma unroll
      for (int c = 0; c < NC; ++c) v = fmaf(xr[c], fs[c][m], v);
      const float w = __expf(v * lss[m] - cmx[m]) * cin[m];
#pragma unroll
      for (int c = 0; c < NC; ++c) acc[c] = fmaf(w, xs[c][m], acc[c]);
    }
  }

  // LayerNorm over C=6 (biased variance, eps=1e-5)
  float mu = 0.0f;
#pragma unroll
  for (int c = 0; c < NC; ++c) mu += acc[c];
  mu *= (1.0f / NC);
  float var = 0.0f;
#pragma unroll
  for (int c = 0; c < NC; ++c) { const float d = acc[c] - mu; var = fmaf(d, d, var); }
  var *= (1.0f / NC);
  const float rs = rsqrtf(var + 1e-5f);
#pragma unroll
  for (int c = 0; c < NC; ++c) {
    const float o = (acc[c] - mu) * rs * norm_w[c] + norm_b[c];
    out[((size_t)(b * NC + c)) * NHW + n] = o;   // [B, C, HW]
  }
}

// ---------------------------------------------------------------------------
extern "C" void kernel_launch(void* const* d_in, const int* in_sizes, int n_in,
                              void* d_out, int out_size, void* d_ws, size_t ws_size,
                              hipStream_t stream) {
  const float* x           = (const float*)d_in[0];  // [8,6,4096]
  const float* feature     = (const float*)d_in[1];  // [8,6,9216]
  const float* fc1_w       = (const float*)d_in[2];  // [9216,9216]
  const float* fc1_b       = (const float*)d_in[3];  // [9216]
  const float* fc2_w       = (const float*)d_in[4];  // [4096,9216]
  const float* fc2_b       = (const float*)d_in[5];  // [4096]
  const float* logit_scale = (const float*)d_in[6];  // [4096]
  const float* norm_w      = (const float*)d_in[7];  // [6]
  const float* norm_b      = (const float*)d_in[8];  // [6]
  float* outp = (float*)d_out;

  float* ws     = (float*)d_ws;
  float* h      = ws;                      // [48, 9216]
  float* f      = h + (size_t)NR * NFF;    // [48, 4096]
  float* colmax = f + (size_t)NR * NHW;    // [8, 4096]
  float* colinv = colmax + (size_t)NB * NHW;
  float* lsv    = colinv + (size_t)NB * NHW;

  // GEMM1 + ReLU: h = relu(feature @ fc1_w^T + fc1_b)
  gemm48_kernel<true><<<NFF / 16, 128, 0, stream>>>(feature, fc1_w, fc1_b, h,
                                                    NFF, NFF);
  // GEMM2: f = h @ fc2_w^T + fc2_b
  gemm48_kernel<false><<<NHW / 16, 128, 0, stream>>>(h, fc2_w, fc2_b, f,
                                                     NFF, NHW);
  // Softmax column stats over n
  colstats_kernel<<<NB * (NHW / 256), 256, 0, stream>>>(x, f, logit_scale,
                                                        colmax, colinv, lsv);
  // Weighted accumulate + LayerNorm + transpose
  attnout_kernel<<<NB * (NHW / 256), 256, 0, stream>>>(x, f, colmax, colinv,
                                                       lsv, norm_w, norm_b,
                                                       outp);
}
